// AnomalyGAT_56727928046358
// MI455X (gfx1250) — matrix operand order repaired and checked by reading, not compiled, not run on hardware
//
#include <hip/hip_runtime.h>
#include <hip/hip_bf16.h>
#include <stdint.h>

typedef __attribute__((ext_vector_type(16))) __bf16        v16bf;
typedef __attribute__((ext_vector_type(8)))  float         v8f;
typedef __attribute__((ext_vector_type(4)))  unsigned int  u32x4;

#define NEG_SLOPE 0.2f
#define CDIV(a, b) (((a) + (b) - 1) / (b))

// ---------- monotone int encoding of float for atomicMax-based segment max ----------
static __device__ __forceinline__ int ord_f32(float f) {
  int i = __float_as_int(f);
  return (i >= 0) ? i : (i ^ 0x7fffffff);
}
static __device__ __forceinline__ float deord_f32(int i) {
  int j = (i >= 0) ? i : (i ^ 0x7fffffff);
  float f = __int_as_float(j);
  // reference: where(isfinite(m), m, 0)
  if (!(f > -3.0e38f && f < 3.0e38f)) f = 0.0f;
  return f;
}

static __device__ __forceinline__ float leaky(float v) {
  return (v > 0.0f) ? v : NEG_SLOPE * v;
}

// ---------- elementwise fp32 -> bf16 ----------
__global__ void k_f32_to_bf16(const float* __restrict__ in,
                              __hip_bfloat16* __restrict__ out, int total) {
  int t = blockIdx.x * blockDim.x + threadIdx.x;
  if (t < total) out[t] = __float2bfloat16(in[t]);
}

__global__ void k_init_int(int* __restrict__ p, int v, int total) {
  int t = blockIdx.x * blockDim.x + threadIdx.x;
  if (t < total) p[t] = v;
}

// ---------- WMMA GEMM: H[n, FOUT] = Xbf[n, FIN] * Wbf[FOUT, FIN]^T (f32 accum) ----------
// W is staged once per block in LDS (cooperative load by all 128 threads), then each
// wave computes a full 16-row strip: one global A-fragment load per k-step feeds
// FOUT/16 WMMAs whose B fragments come from low-latency ds_load_b128.
template <int FIN, int FOUT>
__global__ __launch_bounds__(128, 1)
void k_gemm_strip(const __hip_bfloat16* __restrict__ X,
                  const __hip_bfloat16* __restrict__ Wt,
                  float* __restrict__ H, int n) {
  constexpr int TN = FOUT / 16;
  constexpr int WVEC = FOUT * FIN / 8;          // # of 16B chunks of W
  __shared__ u32x4 sW[WVEC];                    // 32KB (FOUT=128) / 16KB (FOUT=64)

  // cooperative W -> LDS stage (ALL threads, before any early-out)
  const u32x4* wg = (const u32x4*)Wt;
  for (int i = threadIdx.x; i < WVEC; i += 128) sW[i] = wg[i];
  __syncthreads();

  int wave = blockIdx.x * (blockDim.x >> 5) + (threadIdx.x >> 5);
  int lane = threadIdx.x & 31;
  int strips = n >> 4;
  if (wave >= strips) return;  // wave-uniform: EXEC all-1s inside

  int l15 = lane & 15;
  int hi  = lane >> 4;  // 0 for lanes 0-15, 1 for lanes 16-31

  // A: row M = lane&15; lanes 0-15 hold K 0..15, lanes 16-31 hold K 16..31
  // B: col N = lane&15; lanes 0-15 hold K 0..15, lanes 16-31 hold K 16..31 (contiguous k, same order as A)
  const __hip_bfloat16* xrow = X + (size_t)((wave << 4) + l15) * FIN;
  const __hip_bfloat16* wlds = (const __hip_bfloat16*)sW;
  const int koffA = hi ? 16 : 0;
  const int koffB = hi ? 16 : 0;

  v8f acc[TN];
#pragma unroll
  for (int t = 0; t < TN; ++t) acc[t] = v8f{};

#pragma unroll
  for (int k0 = 0; k0 < FIN; k0 += 32) {
    union { v16bf v; u32x4 u[2]; } A;
    const u32x4* pa = (const u32x4*)(xrow + k0 + koffA);
    A.u[0] = pa[0];      // K = kb .. kb+7
    A.u[1] = pa[1];      // K = kb+8 .. kb+15
#pragma unroll
    for (int tn = 0; tn < TN; ++tn) {
      union { v16bf v; u32x4 u[2]; } B;
      const u32x4* pb = (const u32x4*)(wlds + (size_t)((tn << 4) + l15) * FIN + k0 + koffB);
      B.u[0] = pb[0];    // K = kb .. kb+7
      B.u[1] = pb[1];    // K = kb+8 .. kb+15
      acc[tn] = __builtin_amdgcn_wmma_f32_16x16x32_bf16(false, A.v, false, B.v,
                                                        (short)0, acc[tn], false, false);
    }
  }

  // C/D layout: VGPR r -> row (strip*16 + r + hi*8), col (tn*16 + lane&15)
  int rowBase = (wave << 4) + (hi ? 8 : 0);
#pragma unroll
  for (int tn = 0; tn < TN; ++tn) {
    int col = (tn << 4) + l15;
#pragma unroll
    for (int r = 0; r < 8; ++r)
      H[(size_t)(rowBase + r) * FOUT + col] = acc[tn][r];
  }
}

// ---------- per-(node, head) attention logits ----------
__global__ void k_alpha(const float* __restrict__ H, const float* __restrict__ a_src,
                        const float* __restrict__ a_dst, float* __restrict__ as,
                        float* __restrict__ ad, int n, int heads, int ch) {
  int t = blockIdx.x * blockDim.x + threadIdx.x;
  if (t >= n * heads) return;
  int node = t / heads;
  int h = t - node * heads;
  const float* hp = H + (size_t)node * heads * ch + (size_t)h * ch;
  const float* ps = a_src + h * ch;
  const float* pd = a_dst + h * ch;
  float s = 0.0f, d = 0.0f;
  for (int c = 0; c < ch; ++c) { float v = hp[c]; s += v * ps[c]; d += v * pd[c]; }
  as[t] = s;
  ad[t] = d;
}

// ---------- edge pass 1: segment max via ordered-int atomicMax ----------
__global__ void k_edge_max(const int* __restrict__ src, const int* __restrict__ dst,
                           const float* __restrict__ as, const float* __restrict__ ad,
                           int* __restrict__ m, int E, int n, int heads) {
  int e = blockIdx.x * blockDim.x + threadIdx.x;
  if (e >= E + n) return;
  int s = (e < E) ? src[e] : (e - E);   // self loops appended
  int d = (e < E) ? dst[e] : (e - E);
  for (int h = 0; h < heads; ++h) {
    float v = leaky(as[s * heads + h] + ad[d * heads + h]);
    atomicMax(&m[d * heads + h], ord_f32(v));
  }
}

// ---------- edge pass 2: ex = exp(e - max); denom += ex; stash ex ----------
__global__ void k_edge_exp(const int* __restrict__ src, const int* __restrict__ dst,
                           const float* __restrict__ as, const float* __restrict__ ad,
                           const int* __restrict__ m, float* __restrict__ den,
                           float* __restrict__ exbuf, int E, int n, int heads) {
  int e = blockIdx.x * blockDim.x + threadIdx.x;
  if (e >= E + n) return;
  int s = (e < E) ? src[e] : (e - E);
  int d = (e < E) ? dst[e] : (e - E);
  for (int h = 0; h < heads; ++h) {
    float v = leaky(as[s * heads + h] + ad[d * heads + h]);
    float ex = __expf(v - deord_f32(m[d * heads + h]));
    atomicAdd(&den[d * heads + h], ex);
    exbuf[(size_t)e * heads + h] = ex;
  }
}

// ---------- edge pass 2b: alpha = ex / (den + 1e-16) ----------
__global__ void k_edge_norm(const int* __restrict__ dst, const float* __restrict__ den,
                            float* __restrict__ exbuf, int E, int n, int heads) {
  int e = blockIdx.x * blockDim.x + threadIdx.x;
  if (e >= E + n) return;
  int d = (e < E) ? dst[e] : (e - E);
  for (int h = 0; h < heads; ++h)
    exbuf[(size_t)e * heads + h] /= (den[d * heads + h] + 1e-16f);
}

// ---------- edge pass 3: out[dst] += alpha * H[src], one thread per (edge, channel) ----------
__global__ void k_edge_agg(const int* __restrict__ src, const int* __restrict__ dst,
                           const float* __restrict__ H, const float* __restrict__ alpha,
                           float* __restrict__ out, int E, int n, int heads, int ch) {
  long long t = (long long)blockIdx.x * blockDim.x + threadIdx.x;
  int fout = heads * ch;
  long long total = (long long)(E + n) * fout;
  if (t >= total) return;
  int e = (int)(t / fout);
  int c = (int)(t - (long long)e * fout);
  int h = c / ch;
  int s = (e < E) ? src[e] : (e - E);
  int d = (e < E) ? dst[e] : (e - E);
  float a = alpha[(size_t)e * heads + h];
  atomicAdd(&out[(size_t)d * fout + c], a * H[(size_t)s * fout + c]);
}

// ---------- epilogues ----------
__global__ void k_finalize_relu(const float* __restrict__ agg, const float* __restrict__ bias,
                                __hip_bfloat16* __restrict__ xnext, int total, int fout) {
  int t = blockIdx.x * blockDim.x + threadIdx.x;
  if (t >= total) return;
  float v = agg[t] + bias[t % fout];
  xnext[t] = __float2bfloat16(fmaxf(v, 0.0f));
}

__global__ void k_finalize_out(const float* __restrict__ agg, const float* __restrict__ bias,
                               float* __restrict__ y, int total, int fout) {
  int t = blockIdx.x * blockDim.x + threadIdx.x;
  if (t >= total) return;
  y[t] = agg[t] + bias[t % fout];
}

extern "C" void kernel_launch(void* const* d_in, const int* in_sizes, int n_in,
                              void* d_out, int out_size, void* d_ws, size_t ws_size,
                              hipStream_t stream) {
  (void)n_in; (void)out_size; (void)ws_size;
  const float* x  = (const float*)d_in[0];
  const int*   ei = (const int*)d_in[1];
  const int n = in_sizes[0] / 128;   // 100000
  const int E = in_sizes[1] / 2;     // 1600000
  const int* srcI = ei;
  const int* dstI = ei + E;
  const int Etot = E + n;

  // ---- workspace carve (aligned to 256B) ----
  char* w = (char*)d_ws;
  auto take = [&](size_t bytes) { char* p = w; w += (bytes + 255) & ~(size_t)255; return p; };
  float* bufH = (float*)take((size_t)n * 128 * 4);            // GEMM output / layer features
  float* bufO = (float*)take((size_t)n * 128 * 4);            // aggregation output
  char*  xreg = take((size_t)n * 128 * 2);                    // bf16 input features...
  __hip_bfloat16* xbf = (__hip_bfloat16*)xreg;
  float* alphaE = (float*)xreg;                               // ...overlapped with per-edge alpha
                                                              // (dead between gemm and finalize;
                                                              //  needs (E+n)*2*4 = 13.6MB <= 25.6MB)
  __hip_bfloat16* wbf = (__hip_bfloat16*)take(128 * 128 * 2);
  float* asb  = (float*)take((size_t)n * 2 * 4);
  float* adb  = (float*)take((size_t)n * 2 * 4);
  int*   mb   = (int*)take((size_t)n * 2 * 4);
  float* denb = (float*)take((size_t)n * 2 * 4);

  const float* Ws[3] = {(const float*)d_in[2], (const float*)d_in[6],  (const float*)d_in[10]};
  const float* Aa[3] = {(const float*)d_in[3], (const float*)d_in[7],  (const float*)d_in[11]};
  const float* Ab[3] = {(const float*)d_in[4], (const float*)d_in[8],  (const float*)d_in[12]};
  const float* Bb[3] = {(const float*)d_in[5], (const float*)d_in[9],  (const float*)d_in[13]};
  const int fouts[3]  = {128, 128, 64};
  const int headsL[3] = {2, 2, 1};

  const int B = 256;
  const int strips = n / 16;          // 6250 16-row strips, 4 waves / 128-thread block

  // first-layer input to bf16
  k_f32_to_bf16<<<CDIV(n * 128, B), B, 0, stream>>>(x, xbf, n * 128);

  for (int L = 0; L < 3; ++L) {
    const int fin = 128, fout = fouts[L], heads = headsL[L], ch = fout / heads;

    k_f32_to_bf16<<<CDIV(fout * fin, B), B, 0, stream>>>(Ws[L], wbf, fout * fin);

    if (fout == 128)
      k_gemm_strip<128, 128><<<CDIV(strips, 4), 128, 0, stream>>>(xbf, wbf, bufH, n);
    else
      k_gemm_strip<128, 64><<<CDIV(strips, 4), 128, 0, stream>>>(xbf, wbf, bufH, n);

    k_alpha<<<CDIV(n * heads, B), B, 0, stream>>>(bufH, Aa[L], Ab[L], asb, adb, n, heads, ch);

    k_init_int<<<CDIV(n * heads, B), B, 0, stream>>>(mb, (int)0x80000000, n * heads);
    hipMemsetAsync(denb, 0, (size_t)n * heads * 4, stream);
    hipMemsetAsync(bufO, 0, (size_t)n * fout * 4, stream);

    k_edge_max <<<CDIV(Etot, B), B, 0, stream>>>(srcI, dstI, asb, adb, mb, E, n, heads);
    k_edge_exp <<<CDIV(Etot, B), B, 0, stream>>>(srcI, dstI, asb, adb, mb, denb, alphaE, E, n, heads);
    k_edge_norm<<<CDIV(Etot, B), B, 0, stream>>>(dstI, denb, alphaE, E, n, heads);

    long long aggT = (long long)Etot * fout;
    k_edge_agg<<<(int)CDIV(aggT, (long long)B), B, 0, stream>>>(srcI, dstI, bufH, alphaE, bufO,
                                                                E, n, heads, ch);

    if (L < 2) {
      k_finalize_relu<<<CDIV(n * fout, B), B, 0, stream>>>(bufO, Bb[L], xbf, n * fout, fout);
    } else {
      k_finalize_out<<<CDIV(n * fout, B), B, 0, stream>>>(bufO, Bb[L], (float*)d_out, n * fout, fout);
    }
  }
}
